// ViewAwareAttention_50843822850696
// MI455X (gfx1250) — matrix operand order repaired and checked by reading, not compiled, not run on hardware
//
#include <hip/hip_runtime.h>
#include <hip/hip_bf16.h>

// ---------------- problem constants ----------------
#define BB 4
#define NN 1024
#define CC 768
#define HH 12
#define HD 64
#define HID 192
#define TOPK 128
#define SCALE 0.125f     // 1/sqrt(64)

typedef __bf16 bf16_t;
typedef __attribute__((ext_vector_type(16))) __bf16 v16bf;
typedef __attribute__((ext_vector_type(8)))  __bf16 v8bf;
typedef __attribute__((ext_vector_type(8)))  float  v8f;
typedef __attribute__((ext_vector_type(4)))  int    v4i;

// ---------------- CDNA5 async global->LDS path (guarded) ----------------
#if defined(__has_builtin)
#  if __has_builtin(__builtin_amdgcn_global_load_async_to_lds_b128)
#    define HAVE_ASYNC_B128 1
#  endif
#  if __has_builtin(__builtin_amdgcn_s_wait_asynccnt)
#    define HAVE_WAIT_ASYNC 1
#  endif
#endif

typedef __attribute__((address_space(1))) v4i gv4i_t;   // global 16B chunk
typedef __attribute__((address_space(3))) v4i lv4i_t;   // LDS 16B chunk

__device__ __forceinline__ void async_copy16(void* lds_dst, const void* gsrc) {
#ifdef HAVE_ASYNC_B128
  __builtin_amdgcn_global_load_async_to_lds_b128(
      (gv4i_t*)gsrc, (lv4i_t*)lds_dst, 0, 0);
#else
  *(uint4*)lds_dst = *(const uint4*)gsrc;   // fallback: through VGPRs
#endif
}

__device__ __forceinline__ void wait_async() {
#ifdef HAVE_WAIT_ASYNC
  __builtin_amdgcn_s_wait_asynccnt(0);
#elif defined(HAVE_ASYNC_B128)
  asm volatile("s_wait_asynccnt 0" ::: "memory");
#endif
}

// ---------------- WMMA helpers ----------------
__device__ __forceinline__ v8f wmma_bf16(v16bf a, v16bf b, v8f c) {
  return __builtin_amdgcn_wmma_f32_16x16x32_bf16(
      false, a, false, b, (short)0, c, false, false);
}

// A fragment (16x32 bf16, row-major source; tile origin applied).
__device__ __forceinline__ v16bf load_a_frag(const bf16_t* A, int lda) {
  int lane = threadIdx.x & 31;
  int r  = lane & 15;
  int kh = (lane >> 4) << 3;            // 0 or 8
  const bf16_t* p = A + r * lda + kh;
  v8bf lo = *(const v8bf*)(p);
  v8bf hi = *(const v8bf*)(p + 16);
  v16bf a;
#pragma unroll
  for (int e = 0; e < 8; ++e) { a[e] = lo[e]; a[8 + e] = hi[e]; }
  return a;
}

// B fragment (32x16 bf16). BT[n][k] row-major (B transposed), tile origin applied.
__device__ __forceinline__ v16bf load_b_fragT(const bf16_t* BT, int ldb) {
  int lane = threadIdx.x & 31;
  int n  = lane & 15;
  int kb = (lane >> 4) << 3;            // 0 or 8
  const bf16_t* p = BT + n * ldb + kb;
  v8bf lo = *(const v8bf*)(p);
  v8bf hi = *(const v8bf*)(p + 16);
  v16bf b;
#pragma unroll
  for (int e = 0; e < 8; ++e) { b[e] = lo[e]; b[8 + e] = hi[e]; }
  return b;
}

// ---------------- kernel 1: view MLP -> w_nonsemantic ----------------
__global__ __launch_bounds__(256)
void view_mlp_kernel(const float* __restrict__ x,
                     const float* __restrict__ ln_g, const float* __restrict__ ln_b,
                     const float* __restrict__ vw1,  const float* __restrict__ vb1,
                     const float* __restrict__ vw2,  const float* __restrict__ vb2,
                     float* __restrict__ wns) {
  __shared__ float xn[CC];
  __shared__ float hbuf[HID];
  __shared__ float ps[8], ps2[8], logit[3];
  int tok = blockIdx.x;
  int t = threadIdx.x;
  const float* xr = x + (size_t)tok * CC;

  float s = 0.f, s2 = 0.f;
  for (int j = t; j < CC; j += 256) { float v = xr[j]; xn[j] = v; s += v; s2 += v * v; }
#pragma unroll
  for (int off = 16; off > 0; off >>= 1) {
    s  += __shfl_xor(s,  off, 32);
    s2 += __shfl_xor(s2, off, 32);
  }
  if ((t & 31) == 0) { ps[t >> 5] = s; ps2[t >> 5] = s2; }
  __syncthreads();
  float ts = 0.f, ts2 = 0.f;
#pragma unroll
  for (int i = 0; i < 8; ++i) { ts += ps[i]; ts2 += ps2[i]; }
  float mean = ts * (1.0f / CC);
  float var  = ts2 * (1.0f / CC) - mean * mean;
  float inv  = rsqrtf(var + 1e-5f);
  for (int j = t; j < CC; j += 256) xn[j] = (xn[j] - mean) * inv * ln_g[j] + ln_b[j];
  __syncthreads();

  if (t < HID) {
    float acc = vb1[t];
    for (int j = 0; j < CC; ++j) acc += xn[j] * vw1[j * HID + t];
    hbuf[t] = 0.5f * acc * (1.0f + erff(acc * 0.70710678118f));
  }
  __syncthreads();
  if (t < 3) {
    float l = vb2[t];
    for (int j = 0; j < HID; ++j) l += hbuf[j] * vw2[j * 3 + t];
    logit[t] = l;
  }
  __syncthreads();
  if (t == 0) {
    float m = fmaxf(logit[0], fmaxf(logit[1], logit[2]));
    float e0 = __expf(logit[0] - m), e1 = __expf(logit[1] - m), e2 = __expf(logit[2] - m);
    wns[tok] = (e1 + e2) / (e0 + e1 + e2);
  }
}

// ---------------- conversion / transpose kernels ----------------
__global__ __launch_bounds__(256)
void convert_bf16_kernel(const float* __restrict__ src, bf16_t* __restrict__ dst, int n) {
  int i = blockIdx.x * 256 + threadIdx.x;
  if (i < n) dst[i] = (bf16_t)src[i];
}

__global__ __launch_bounds__(256)
void transpose_bf16_kernel(const float* __restrict__ src, bf16_t* __restrict__ dst,
                           int R, int Cc) {
  int i = blockIdx.x * 256 + threadIdx.x;
  if (i < R * Cc) {
    int r = i / Cc, c = i % Cc;
    dst[(size_t)c * R + r] = (bf16_t)src[i];
  }
}

// ---------------- kernel 2: QKV GEMM, 32x64 register-blocked ----------------
__global__ __launch_bounds__(256)
void gemm_qkv_kernel(const bf16_t* __restrict__ xbf, const bf16_t* __restrict__ wT,
                     bf16_t* __restrict__ qb, bf16_t* __restrict__ kb,
                     bf16_t* __restrict__ vb) {
  int wid  = (blockIdx.x * 256 + threadIdx.x) >> 5;
  int lane = threadIdx.x & 31;
  int tm = wid & 127;                    // 4096/32 = 128 row tiles
  int tn = wid >> 7;                     // 2304/64 = 36 col tiles
  int m0 = tm << 5, n0 = tn << 6;
  v8f zero = {0.f, 0.f, 0.f, 0.f, 0.f, 0.f, 0.f, 0.f};
  v8f acc[2][4];
#pragma unroll
  for (int i = 0; i < 2; ++i)
#pragma unroll
    for (int j = 0; j < 4; ++j) acc[i][j] = zero;

#pragma unroll 2
  for (int k0 = 0; k0 < CC; k0 += 32) {
    __builtin_prefetch(xbf + (size_t)m0 * CC + k0 + 64, 0, 1);
    v16bf a0 = load_a_frag(xbf + (size_t)(m0 +  0) * CC + k0, CC);
    v16bf a1 = load_a_frag(xbf + (size_t)(m0 + 16) * CC + k0, CC);
#pragma unroll
    for (int j = 0; j < 4; ++j) {
      v16bf bb = load_b_fragT(wT + (size_t)(n0 + 16 * j) * CC + k0, CC);
      acc[0][j] = wmma_bf16(a0, bb, acc[0][j]);
      acc[1][j] = wmma_bf16(a1, bb, acc[1][j]);
    }
  }

  int col = lane & 15, rbase = (lane >> 4) << 3;
#pragma unroll
  for (int jt = 0; jt < 4; ++jt) {
    int j  = n0 + jt * 16 + col;
    int t3 = j / CC;
    int jj = j % CC;
    int h = jj >> 6, d = jj & 63;
#pragma unroll
    for (int it = 0; it < 2; ++it) {
#pragma unroll
      for (int i = 0; i < 8; ++i) {
        int m   = m0 + it * 16 + rbase + i;
        int b_  = m >> 10;
        int tok = m & 1023;
        bf16_t val = (bf16_t)acc[it][jt][i];
        if (t3 == 0)      qb[(((size_t)b_ * HH + h) * NN + tok) * HD + d] = val;
        else if (t3 == 1) kb[(((size_t)b_ * HH + h) * NN + tok) * HD + d] = val;
        else              vb[(((size_t)b_ * HH + h) * HD + d) * NN + tok] = val; // V^T
      }
    }
  }
}

// ---------------- kernel 3: attention -------------------------------------
// Block = 4 waves = 4 consecutive 16-query tiles of one (b,h).
// LDS: 128KB shared K-panel (recycled as V^T-panel) + 4 x 32KB bf16 score rows.
#define ATT_WAVES 4
#define PANEL_ELEMS (NN * HD)                 // 65536 bf16 = 128KB
__global__ __launch_bounds__(ATT_WAVES * 32)
void attn_kernel(const bf16_t* __restrict__ q, const bf16_t* __restrict__ k,
                 const bf16_t* __restrict__ vT, const float* __restrict__ wns,
                 bf16_t* __restrict__ obf) {
  extern __shared__ char smem_raw[];
  bf16_t* panel = (bf16_t*)smem_raw;                         // K then V^T
  bf16_t* Sall  = (bf16_t*)(smem_raw + PANEL_ELEMS * 2);     // 4 x 16 x 1024 bf16
  float*  rsall = (float*)(smem_raw + PANEL_ELEMS * 2 + ATT_WAVES * 16 * NN * 2);

  int tid  = threadIdx.x;
  int wave = tid >> 5;
  int lane = tid & 31;
  bf16_t* S  = Sall + (size_t)wave * 16 * NN;
  float*  rs = rsall + wave * 16;

  int bh    = blockIdx.x >> 4;            // 16 blocks per (b,h)
  int tile4 = blockIdx.x & 15;
  int b = bh / HH, h = bh % HH;
  int r0 = (tile4 * 4 + wave) << 4;

  const bf16_t* Q = q  + (size_t)bh * NN * HD;
  const bf16_t* K = k  + (size_t)bh * NN * HD;
  const bf16_t* V = vT + (size_t)bh * HD * NN;
  const float* wrow = wns + (size_t)b * NN;

  // ---- stage K panel (128KB) into LDS, cooperatively, async ----
  for (int c = tid; c < PANEL_ELEMS / 8; c += ATT_WAVES * 32)
    async_copy16(panel + (size_t)c * 8, K + (size_t)c * 8);
  wait_async();
  __syncthreads();

  int col = lane & 15, rbase = (lane >> 4) << 3;

  // ---- phase 1: S = Q K^T * scale + bias(key); Q frags hoisted ----
  v16bf aq0 = load_a_frag(Q + (size_t)r0 * HD +  0, HD);
  v16bf aq1 = load_a_frag(Q + (size_t)r0 * HD + 32, HD);
  for (int kt = 0; kt < NN / 16; ++kt) {
    v8f acc = {0.f, 0.f, 0.f, 0.f, 0.f, 0.f, 0.f, 0.f};
    v16bf b0 = load_b_fragT(panel + (size_t)(kt * 16) * HD +  0, HD);   // LDS
    v16bf b1 = load_b_fragT(panel + (size_t)(kt * 16) * HD + 32, HD);   // LDS
    acc = wmma_bf16(aq0, b0, acc);
    acc = wmma_bf16(aq1, b1, acc);
    int key = (kt << 4) + col;
    float bias = wrow[key];                              // ALPHA = 1
#pragma unroll
    for (int i = 0; i < 8; ++i)
      S[(rbase + i) * NN + key] = (bf16_t)(acc[i] * SCALE + bias);
  }
  __syncthreads();   // everyone done reading K panel

  // ---- kick off async V^T staging; overlaps with top-k phase ----
  for (int c = tid; c < PANEL_ELEMS / 8; c += ATT_WAVES * 32)
    async_copy16(panel + (size_t)c * 8, V + (size_t)c * 8);

  // ---- phase 2: per-row top-128 threshold + masked exp ----
  for (int r = 0; r < 16; ++r) {
    bf16_t* row = S + r * NN;
    float mx = -3.4e38f, mn = 3.4e38f;
    for (int j = lane; j < NN; j += 32) {
      float s = (float)row[j];
      mx = fmaxf(mx, s); mn = fminf(mn, s);
    }
#pragma unroll
    for (int off = 16; off > 0; off >>= 1) {
      mx = fmaxf(mx, __shfl_xor(mx, off, 32));
      mn = fminf(mn, __shfl_xor(mn, off, 32));
    }
    float lo = mn, hi = mx;
#pragma unroll 1
    for (int it = 0; it < 24; ++it) {                    // binary-search kth value
      float mid = 0.5f * (lo + hi);
      int cnt = 0;
      for (int j = lane; j < NN; j += 32) cnt += ((float)row[j] >= mid) ? 1 : 0;
#pragma unroll
      for (int off = 16; off > 0; off >>= 1) cnt += __shfl_xor(cnt, off, 32);
      if (cnt >= TOPK) lo = mid; else hi = mid;
    }
    float thr = lo;
    float sum = 0.f;
    for (int j = lane; j < NN; j += 32) {
      float s = (float)row[j];
      float e = (s >= thr) ? __expf(s - mx) : 0.0f;
      row[j] = (bf16_t)e;
      sum += e;
    }
#pragma unroll
    for (int off = 16; off > 0; off >>= 1) sum += __shfl_xor(sum, off, 32);
    if (lane == 0) rs[r] = 1.0f / sum;
  }
  wait_async();
  __syncthreads();   // V^T panel ready, S finalized

  // ---- phase 3: O = P V (both operands from LDS) ----
  for (int n0 = 0; n0 < HD; n0 += 16) {
    v8f acc = {0.f, 0.f, 0.f, 0.f, 0.f, 0.f, 0.f, 0.f};
#pragma unroll 4
    for (int kv0 = 0; kv0 < NN; kv0 += 32) {
      v16bf a  = load_a_frag(S + kv0, NN);                        // LDS bf16
      v16bf bb = load_b_fragT(panel + (size_t)n0 * NN + kv0, NN); // LDS V^T
      acc = wmma_bf16(a, bb, acc);
    }
#pragma unroll
    for (int i = 0; i < 8; ++i) {
      int r   = rbase + i;
      int tok = r0 + r;
      float val = acc[i] * rs[r];
      obf[((size_t)b * NN + tok) * CC + h * HD + n0 + col] = (bf16_t)val;
    }
  }
}

// ---------------- kernel 4: output projection, 32x64 register-blocked ------
__global__ __launch_bounds__(256)
void gemm_proj_kernel(const bf16_t* __restrict__ obf, const bf16_t* __restrict__ pwT,
                      const float* __restrict__ pb, float* __restrict__ out) {
  int wid  = (blockIdx.x * 256 + threadIdx.x) >> 5;
  int lane = threadIdx.x & 31;
  int tm = wid & 127;                    // 128 row tiles
  int tn = wid >> 7;                     // 768/64 = 12 col tiles
  int m0 = tm << 5, n0 = tn << 6;
  v8f zero = {0.f, 0.f, 0.f, 0.f, 0.f, 0.f, 0.f, 0.f};
  v8f acc[2][4];
#pragma unroll
  for (int i = 0; i < 2; ++i)
#pragma unroll
    for (int j = 0; j < 4; ++j) acc[i][j] = zero;

#pragma unroll 2
  for (int k0 = 0; k0 < CC; k0 += 32) {
    __builtin_prefetch(obf + (size_t)m0 * CC + k0 + 64, 0, 1);
    v16bf a0 = load_a_frag(obf + (size_t)(m0 +  0) * CC + k0, CC);
    v16bf a1 = load_a_frag(obf + (size_t)(m0 + 16) * CC + k0, CC);
#pragma unroll
    for (int j = 0; j < 4; ++j) {
      v16bf bb = load_b_fragT(pwT + (size_t)(n0 + 16 * j) * CC + k0, CC);
      acc[0][j] = wmma_bf16(a0, bb, acc[0][j]);
      acc[1][j] = wmma_bf16(a1, bb, acc[1][j]);
    }
  }

  int col = lane & 15, rbase = (lane >> 4) << 3;
#pragma unroll
  for (int jt = 0; jt < 4; ++jt) {
    int j = n0 + jt * 16 + col;
    float bias = pb[j];
#pragma unroll
    for (int it = 0; it < 2; ++it) {
#pragma unroll
      for (int i = 0; i < 8; ++i) {
        int m = m0 + it * 16 + rbase + i;
        out[(size_t)m * CC + j] = acc[it][jt][i] + bias;
      }
    }
  }
}

// ---------------- host launcher ----------------
extern "C" void kernel_launch(void* const* d_in, const int* in_sizes, int n_in,
                              void* d_out, int out_size, void* d_ws, size_t ws_size,
                              hipStream_t stream) {
  const float* x      = (const float*)d_in[0];
  const float* qkv_w  = (const float*)d_in[1];
  const float* proj_w = (const float*)d_in[2];
  const float* proj_b = (const float*)d_in[3];
  const float* ln_g   = (const float*)d_in[4];
  const float* ln_b   = (const float*)d_in[5];
  const float* vw1    = (const float*)d_in[6];
  const float* vb1    = (const float*)d_in[7];
  const float* vw2    = (const float*)d_in[8];
  const float* vb2    = (const float*)d_in[9];
  float* out = (float*)d_out;

  char* ws = (char*)d_ws;
  const size_t OFF_WNS    = 0;                         // 4096 f32
  const size_t OFF_XBF    = 16384;                     // 4096*768 bf16
  const size_t OFF_QKVWT  = OFF_XBF   + 6291456;       // 2304*768 bf16
  const size_t OFF_PROJWT = OFF_QKVWT + 3538944;       // 768*768 bf16
  const size_t OFF_QBF    = OFF_PROJWT + 1179648;      // B*H*N*64 bf16
  const size_t OFF_KBF    = OFF_QBF   + 6291456;
  const size_t OFF_VBF    = OFF_KBF   + 6291456;
  const size_t OFF_OBF    = OFF_VBF   + 6291456;

  float*  wns    = (float*)(ws + OFF_WNS);
  bf16_t* x_bf   = (bf16_t*)(ws + OFF_XBF);
  bf16_t* qkv_wT = (bf16_t*)(ws + OFF_QKVWT);
  bf16_t* projWT = (bf16_t*)(ws + OFF_PROJWT);
  bf16_t* q_bf   = (bf16_t*)(ws + OFF_QBF);
  bf16_t* k_bf   = (bf16_t*)(ws + OFF_KBF);
  bf16_t* v_bf   = (bf16_t*)(ws + OFF_VBF);
  bf16_t* o_bf   = (bf16_t*)(ws + OFF_OBF);

  // LDS: 128KB panel + 128KB scores + row-sum scalars
  const int attn_smem = PANEL_ELEMS * 2 + ATT_WAVES * 16 * NN * 2 + ATT_WAVES * 16 * 4;
  (void)hipFuncSetAttribute((const void*)attn_kernel,
                            hipFuncAttributeMaxDynamicSharedMemorySize, attn_smem);

  // 1) view MLP -> per-key bias weights
  view_mlp_kernel<<<BB * NN, 256, 0, stream>>>(x, ln_g, ln_b, vw1, vb1, vw2, vb2, wns);

  // 2) precision prep
  convert_bf16_kernel<<<(BB * NN * CC + 255) / 256, 256, 0, stream>>>(x, x_bf, BB * NN * CC);
  transpose_bf16_kernel<<<(CC * 3 * CC + 255) / 256, 256, 0, stream>>>(qkv_w, qkv_wT, CC, 3 * CC);
  transpose_bf16_kernel<<<(CC * CC + 255) / 256, 256, 0, stream>>>(proj_w, projWT, CC, CC);

  // 3) QKV projection: 128 x 36 wave tiles (32x64 each), 8 waves/block
  gemm_qkv_kernel<<<(128 * 36) / 8, 256, 0, stream>>>(x_bf, qkv_wT, q_bf, k_bf, v_bf);

  // 4) attention: B*H*(N/16) = 3072 tiles, 4 waves/block sharing K/V panels
  attn_kernel<<<3072 / ATT_WAVES, ATT_WAVES * 32, attn_smem, stream>>>(q_bf, k_bf, v_bf, wns, o_bf);

  // 5) output projection: 128 x 12 wave tiles
  gemm_proj_kernel<<<(128 * 12) / 8, 256, 0, stream>>>(o_bf, projWT, proj_b, out);
}
